// GraphEncode_85452669321959
// MI455X (gfx1250) — matrix operand order repaired and checked by reading, not compiled, not run on hardware
//
#include <hip/hip_runtime.h>
#include <hip/hip_bf16.h>

typedef __attribute__((ext_vector_type(16))) __bf16 v16bf;
typedef __attribute__((ext_vector_type(8)))  float  v8f;

constexpr int BG  = 64;            // graphs
constexpr int NN  = 1024;          // nodes per graph
constexpr int DEG = 32;
constexpr int CH  = 128;           // IN_CH == HID
constexpr int NB  = BG * NN;       // 65536 nodes total
constexpr int NE  = BG * NN * DEG; // 2097152 edges
constexpr int KK  = NN / 2;        // 512 kept per graph
constexpr int M2  = BG * KK;       // 32768 pooled nodes

// cold-path conversion (explicit RNE)
__device__ __forceinline__ unsigned short f2bf(float f) {
    unsigned u = __builtin_bit_cast(unsigned, f);
    u += 0x7FFFu + ((u >> 16) & 1u);
    return (unsigned short)(u >> 16);
}
// hot-path conversion: let clang use the native f32->bf16 convert
__device__ __forceinline__ unsigned short f2bf_hw(float f) {
    __bf16 h = (__bf16)f;
    return __builtin_bit_cast(unsigned short, h);
}
__device__ __forceinline__ __bf16 us2bf(unsigned short u) {
    return __builtin_bit_cast(__bf16, u);
}

// ---------------- utility kernels ----------------
__global__ void k_zero(float* p, int n) {
    int i = blockIdx.x * blockDim.x + threadIdx.x;
    int stride = gridDim.x * blockDim.x;
    for (; i < n; i += stride) p[i] = 0.0f;
}

// Pack a 128x128 fp32 weight (row = K, col = N) into WMMA B-fragment order:
// packed[(((ntile*4 + kc)*32 + lane)*16) + e] = bf16( W[k*CH + col] )
//   col = ntile*16 + lane%16 ; half = lane/16 ; k = kc*32 + (e<8 ? half*8+e : 16+half*8+(e-8))
// so each lane's 16-element fragment is 32 contiguous bytes -> 2x global_load_b128.
__global__ void k_pack_w(const float* __restrict__ W, unsigned short* __restrict__ P) {
    int i = blockIdx.x * blockDim.x + threadIdx.x;   // 0 .. 16383
    if (i >= CH * CH) return;
    int e    = i & 15;
    int lane = (i >> 4) & 31;
    int kc   = (i >> 9) & 3;
    int n    = i >> 11;
    int col  = n * 16 + (lane & 15);
    int half = lane >> 4;
    int k    = kc * 32 + (e < 8 ? half * 8 + e : 16 + half * 8 + (e - 8));
    P[i] = f2bf(W[k * CH + col]);
}

// score = tanh((x @ w_pool) / ||w_pool||), per node
__global__ void k_score(const float* __restrict__ x, const float* __restrict__ wp,
                        float* __restrict__ score) {
    int node = blockIdx.x * blockDim.x + threadIdx.x;
    if (node >= NB) return;
    const float* xr = x + (long)node * CH;
    float nrm = 0.f, dot = 0.f;
    #pragma unroll 8
    for (int c = 0; c < CH; ++c) { float w = wp[c]; nrm += w * w; dot += xr[c] * w; }
    score[node] = tanhf(dot * rsqrtf(nrm));
}

// per-graph exact top-K by rank counting (matches lax.top_k tie-break: lower index wins)
__global__ void k_topk(const float* __restrict__ score, int* __restrict__ o2n,
                       int* __restrict__ n2o, float* __restrict__ gate) {
    __shared__ float s[NN];
    int g = blockIdx.x, t = threadIdx.x;
    s[t] = score[g * NN + t];
    __syncthreads();
    float mine = s[t];
    int rank = 0;
    for (int j = 0; j < NN; ++j) {
        float v = s[j];
        rank += (v > mine) || (v == mine && j < t);
    }
    int old = g * NN + t;
    if (rank < KK) {
        int nw = g * KK + rank;
        o2n[old] = nw;
        n2o[nw]  = old;
        gate[nw] = mine;
    } else {
        o2n[old] = -1;
    }
}

// conv1 aggregation: agg[dst] += x[src] over all edges (128 threads = channels)
__global__ void k_agg1(const float* __restrict__ x, const int* __restrict__ src,
                       const int* __restrict__ dst, float* __restrict__ agg) {
    int c = threadIdx.x;
    int e0 = blockIdx.x * 8;
    #pragma unroll
    for (int k = 0; k < 8; ++k) {
        int e = e0 + k;
        int s = src[e], d = dst[e];
        atomicAdd(&agg[(long)d * CH + c], x[(long)s * CH + c]);
    }
}

// conv2 aggregation on the pooled graph: only edges whose endpoints survived
__global__ void k_agg2(const float* __restrict__ pooled, const int* __restrict__ src,
                       const int* __restrict__ dst, const int* __restrict__ o2n,
                       float* __restrict__ agg) {
    int c = threadIdx.x;
    int e0 = blockIdx.x * 8;
    #pragma unroll
    for (int k = 0; k < 8; ++k) {
        int e = e0 + k;
        int s2 = o2n[src[e]], d2 = o2n[dst[e]];
        if (s2 >= 0 && d2 >= 0)
            atomicAdd(&agg[(long)d2 * CH + c], pooled[(long)s2 * CH + c]);
    }
}

// Fused GIN MLP: out = relu( relu((agg+base) @ Wa + ba) @ Wb + bb )
// 256 threads = 8 waves; block computes a 16-row tile; wave w owns cols [16w,16w+16).
// Wa/Wb are pre-packed B-fragments (see k_pack_w).
__global__ void k_gin_mlp(const float* __restrict__ agg, const float* __restrict__ base,
                          const v16bf* __restrict__ Wa, const float* __restrict__ ba,
                          const v16bf* __restrict__ Wb, const float* __restrict__ bb,
                          float* __restrict__ out) {
    __shared__ unsigned short A0[16][CH];
    __shared__ unsigned short A1[16][CH];
    const int tid = threadIdx.x;
    const long rowbase = (long)blockIdx.x * 16;

    // stage h = agg + base as bf16 into LDS (vectorized: float4 in, ushort4 out)
    for (int i = tid * 4; i < 16 * CH; i += 256 * 4) {
        int r = i >> 7, c = i & 127;
        long idx = (rowbase + r) * CH + c;
        float4 va = *(const float4*)(agg + idx);
        float4 vb = *(const float4*)(base + idx);
        ushort4 u;
        u.x = f2bf_hw(va.x + vb.x);
        u.y = f2bf_hw(va.y + vb.y);
        u.z = f2bf_hw(va.z + vb.z);
        u.w = f2bf_hw(va.w + vb.w);
        *(ushort4*)&A0[r][c] = u;
    }
    __syncthreads();

    const int wave = tid >> 5;
    const int lane = tid & 31;
    const int half = lane >> 4;               // lane group 0-15 / 16-31
    const int row  = lane & 15;               // A-matrix row (M)
    const int col  = wave * 16 + (lane & 15); // output column (N)

    // ---- GEMM 1: (16x128) @ Wa(128x128), this wave's 16 cols ----
    v8f acc = {0.f, 0.f, 0.f, 0.f, 0.f, 0.f, 0.f, 0.f};
    #pragma unroll
    for (int kc = 0; kc < 4; ++kc) {
        v16bf a;
        #pragma unroll
        for (int j = 0; j < 8; ++j) {    // A layout: VGPR0-3 K=half*8+j, VGPR4-7 K=16+half*8+j
            int k1 = kc * 32 + half * 8 + j;
            a[j]     = us2bf(A0[row][k1]);
            a[j + 8] = us2bf(A0[row][k1 + 16]);
        }
        v16bf b = Wa[(wave * 4 + kc) * 32 + lane];   // pre-packed: 32 contiguous bytes/lane
        acc = __builtin_amdgcn_wmma_f32_16x16x32_bf16(false, a, false, b,
                                                      (short)0, acc, false, false);
    }
    float bias_a = ba[col];
    #pragma unroll
    for (int r = 0; r < 8; ++r) {        // C layout: M = r + 8*half, N = col
        int m = r + 8 * half;
        float v = acc[r] + bias_a;
        v = v > 0.f ? v : 0.f;           // inner relu
        A1[m][col] = f2bf_hw(v);
    }
    __syncthreads();

    // ---- GEMM 2: relu-out @ Wb ----
    v8f acc2 = {0.f, 0.f, 0.f, 0.f, 0.f, 0.f, 0.f, 0.f};
    #pragma unroll
    for (int kc = 0; kc < 4; ++kc) {
        v16bf a;
        #pragma unroll
        for (int j = 0; j < 8; ++j) {
            int k1 = kc * 32 + half * 8 + j;
            a[j]     = us2bf(A1[row][k1]);
            a[j + 8] = us2bf(A1[row][k1 + 16]);
        }
        v16bf b = Wb[(wave * 4 + kc) * 32 + lane];
        acc2 = __builtin_amdgcn_wmma_f32_16x16x32_bf16(false, a, false, b,
                                                       (short)0, acc2, false, false);
    }
    float bias_b = bb[col];
    #pragma unroll
    for (int r = 0; r < 8; ++r) {
        int m = r + 8 * half;
        float v = acc2[r] + bias_b;
        v = v > 0.f ? v : 0.f;           // outer relu (both convs apply it)
        out[(rowbase + m) * CH + col] = v;
    }
}

// pooled[new] = out1[old(new)] * gate[new]
__global__ void k_pool(const float* __restrict__ h1, const int* __restrict__ n2o,
                       const float* __restrict__ gate, float* __restrict__ pooled) {
    int i = blockIdx.x * blockDim.x + threadIdx.x;
    if (i >= M2 * CH) return;
    int node = i >> 7, c = i & 127;
    pooled[i] = h1[(long)n2o[node] * CH + c] * gate[node];
}

// per-graph mean over the K kept nodes -> (B, CH)
__global__ void k_mean(const float* __restrict__ h2, float* __restrict__ out) {
    int b = blockIdx.x, c = threadIdx.x;
    float s = 0.f;
    const float* p = h2 + (long)b * KK * CH + c;
    for (int k = 0; k < KK; ++k) s += p[(long)k * CH];
    out[b * CH + c] = s * (1.0f / KK);
}

extern "C" void kernel_launch(void* const* d_in, const int* in_sizes, int n_in,
                              void* d_out, int out_size, void* d_ws, size_t ws_size,
                              hipStream_t stream) {
    const float* x     = (const float*)d_in[0];
    const int*   src   = (const int*)d_in[1];         // edge_index row 0
    const int*   dst   = src + NE;                    // edge_index row 1
    const float* W1    = (const float*)d_in[3];
    const float* b1    = (const float*)d_in[4];
    const float* W2    = (const float*)d_in[5];
    const float* b2    = (const float*)d_in[6];
    const float* wpool = (const float*)d_in[7];
    const float* W3    = (const float*)d_in[8];
    const float* b3    = (const float*)d_in[9];
    const float* W4    = (const float*)d_in[10];
    const float* b4    = (const float*)d_in[11];
    float* out = (float*)d_out;

    // workspace carve-up
    char* ws = (char*)d_ws;
    float* h1     = (float*)ws; ws += (size_t)NB * CH * 4;   // 32 MB: agg1, then MLP1 output (aliased)
    float* h2     = (float*)ws; ws += (size_t)M2 * CH * 4;   // 16 MB: agg2, then MLP2 output (aliased)
    float* pooled = (float*)ws; ws += (size_t)M2 * CH * 4;   // 16 MB
    float* score  = (float*)ws; ws += (size_t)NB * 4;
    int*   o2n    = (int*)ws;   ws += (size_t)NB * 4;
    int*   n2o    = (int*)ws;   ws += (size_t)M2 * 4;
    float* gate   = (float*)ws; ws += (size_t)M2 * 4;
    unsigned short* Wpk = (unsigned short*)ws;               // 4 x 128x128 bf16, fragment order
    unsigned short* W1p = Wpk;
    unsigned short* W2p = Wpk + 1 * CH * CH;
    unsigned short* W3p = Wpk + 2 * CH * CH;
    unsigned short* W4p = Wpk + 3 * CH * CH;

    // weights -> bf16 WMMA-fragment layout, once per launch
    k_pack_w<<<(CH * CH + 255) / 256, 256, 0, stream>>>(W1, W1p);
    k_pack_w<<<(CH * CH + 255) / 256, 256, 0, stream>>>(W2, W2p);
    k_pack_w<<<(CH * CH + 255) / 256, 256, 0, stream>>>(W3, W3p);
    k_pack_w<<<(CH * CH + 255) / 256, 256, 0, stream>>>(W4, W4p);

    // zero the two segment-sum accumulators
    k_zero<<<4096, 256, 0, stream>>>(h1, NB * CH);
    k_zero<<<4096, 256, 0, stream>>>(h2, M2 * CH);

    // TopK scores + per-graph selection (independent of conv1 path)
    k_score<<<NB / 256, 256, 0, stream>>>(x, wpool, score);
    k_topk<<<BG, NN, 0, stream>>>(score, o2n, n2o, gate);

    // conv1: aggregate + fused WMMA MLP
    k_agg1<<<NE / 8, 128, 0, stream>>>(x, src, dst, h1);
    k_gin_mlp<<<NB / 16, 256, 0, stream>>>(h1, x, (const v16bf*)W1p, b1,
                                           (const v16bf*)W2p, b2, h1);

    // TopK gate/gather
    k_pool<<<(M2 * CH) / 256, 256, 0, stream>>>(h1, n2o, gate, pooled);

    // conv2 on pooled graph
    k_agg2<<<NE / 8, 128, 0, stream>>>(pooled, src, dst, o2n, h2);
    k_gin_mlp<<<M2 / 16, 256, 0, stream>>>(h2, pooled, (const v16bf*)W3p, b3,
                                           (const v16bf*)W4p, b4, h2);

    // global mean pool
    k_mean<<<BG, CH, 0, stream>>>(h2, out);
}
